// SIGN_47107201303134
// MI455X (gfx1250) — compile-verified, matrix operand order checked by reading, not run on hardware
//
#include <hip/hip_runtime.h>
#include <hip/hip_bf16.h>
#include <stdint.h>

#define NN    100000
#define NE    3200000
#define INC   256
#define HIDC  256
#define OUTC  128
#define CAT   1024           // IN_C * (HOPS+1)
#define TILE_P 264           // padded LDS row stride (bf16) for transpose tiles
#define SLICE_P 40           // padded LDS row stride (bf16) for weight K-slices (80B: bank-conflict-free)

typedef __attribute__((ext_vector_type(16))) __bf16 v16bf;
typedef __attribute__((ext_vector_type(8)))  __bf16 v8bf;
typedef __attribute__((ext_vector_type(8)))  float  v8f;

// ---------------- preprocessing kernels ----------------

__global__ void init_xc_kernel(const float* __restrict__ x, float* __restrict__ xc,
                               long long total) {
    long long i = (long long)blockIdx.x * blockDim.x + threadIdx.x;
    if (i >= total) return;
    int c = (int)(i & (CAT - 1));
    long long r = i >> 10;
    xc[i] = (c < INC) ? x[r * INC + c] : 0.0f;   // block 0 = x, hop blocks zeroed
}

__global__ void zero_f32_kernel(float* __restrict__ p, int n) {
    int i = blockIdx.x * blockDim.x + threadIdx.x;
    if (i < n) p[i] = 0.0f;
}

__global__ void degree_kernel(const long long* __restrict__ row,
                              float* __restrict__ deg, int e) {
    int i = blockIdx.x * blockDim.x + threadIdx.x;
    if (i < e) atomicAdd(&deg[(int)row[i]], 1.0f);
}

__global__ void rsqrt_kernel(float* __restrict__ deg, int n) {
    int i = blockIdx.x * blockDim.x + threadIdx.x;
    if (i < n) {
        float d = deg[i];
        deg[i] = (d > 0.0f) ? rsqrtf(d) : 0.0f;
    }
}

__global__ void edge_value_kernel(const long long* __restrict__ row,
                                  const long long* __restrict__ col,
                                  const float* __restrict__ dinv,
                                  float* __restrict__ val, int e) {
    int i = blockIdx.x * blockDim.x + threadIdx.x;
    if (i < e) val[i] = dinv[(int)row[i]] * dinv[(int)col[i]];
}

__global__ void cvt_bf16_kernel(const float* __restrict__ s, __bf16* __restrict__ d, int n) {
    int i = blockIdx.x * blockDim.x + threadIdx.x;
    if (i < n) d[i] = (__bf16)s[i];
}

// one wave per edge, 8 channels per lane; gather col row, scatter-atomic into row row
__global__ void spmm_kernel(const long long* __restrict__ row,
                            const long long* __restrict__ col,
                            const float* __restrict__ val,
                            float* __restrict__ xc,
                            int src_off, int dst_off, int e) {
    int gid  = blockIdx.x * blockDim.x + threadIdx.x;
    int edge = gid >> 5;
    int lane = gid & 31;
    if (edge >= e) return;
    long long r = row[edge];
    long long c = col[edge];
    float v = val[edge];
    const float* sp = xc + c * (long long)CAT + src_off + lane * 8;
    float4 s0 = *(const float4*)sp;
    float4 s1 = *(const float4*)(sp + 4);
    float* dp = xc + r * (long long)CAT + dst_off + lane * 8;
    atomicAdd(dp + 0, v * s0.x);
    atomicAdd(dp + 1, v * s0.y);
    atomicAdd(dp + 2, v * s0.z);
    atomicAdd(dp + 3, v * s0.w);
    atomicAdd(dp + 4, v * s1.x);
    atomicAdd(dp + 5, v * s1.y);
    atomicAdd(dp + 6, v * s1.z);
    atomicAdd(dp + 7, v * s1.w);
}

// ---------------- fused WMMA MLP ----------------
// A fragment (16x32 bf16): lane holds row M=lane&15; elems 0..7 = K kOff..kOff+7,
// elems 8..15 = K kOff+16..kOff+23, where kOff = kBase + 8*(lane>>4).
__device__ __forceinline__ v16bf load_a_f32(const float* __restrict__ rowp, int kOff) {
    float4 a0 = *(const float4*)(rowp + kOff);
    float4 a1 = *(const float4*)(rowp + kOff + 4);
    float4 a2 = *(const float4*)(rowp + kOff + 16);
    float4 a3 = *(const float4*)(rowp + kOff + 20);
    v16bf r;
    r[0]  = (__bf16)a0.x; r[1]  = (__bf16)a0.y; r[2]  = (__bf16)a0.z; r[3]  = (__bf16)a0.w;
    r[4]  = (__bf16)a1.x; r[5]  = (__bf16)a1.y; r[6]  = (__bf16)a1.z; r[7]  = (__bf16)a1.w;
    r[8]  = (__bf16)a2.x; r[9]  = (__bf16)a2.y; r[10] = (__bf16)a2.z; r[11] = (__bf16)a2.w;
    r[12] = (__bf16)a3.x; r[13] = (__bf16)a3.y; r[14] = (__bf16)a3.z; r[15] = (__bf16)a3.w;
    return r;
}

__device__ __forceinline__ v16bf load_frag_bf16(const __bf16* p) {
    v8bf lo = *(const v8bf*)p;
    v8bf hi = *(const v8bf*)(p + 16);
    v16bf r;
#pragma unroll
    for (int j = 0; j < 8; ++j) { r[j] = lo[j]; r[j + 8] = hi[j]; }
    return r;
}

// One K-step of NTILES WMMAs with depth-2 software-pipelined B-fragment LDS loads.
// bsw must already be offset by this wave's n-range (nbase rows).
template <int NTILES>
__device__ __forceinline__ void wmma_kstep(const __bf16* bsw, int ln, int half,
                                           v16bf a, v8f (&acc)[8]) {
    v16bf bb[3];
    bb[0] = load_frag_bf16(bsw + (0 * 16 + ln) * SLICE_P + half * 8);
    bb[1] = load_frag_bf16(bsw + (1 * 16 + ln) * SLICE_P + half * 8);
#pragma unroll
    for (int nt = 0; nt < NTILES; ++nt) {
        if (nt + 2 < NTILES)
            bb[(nt + 2) % 3] = load_frag_bf16(bsw + ((nt + 2) * 16 + ln) * SLICE_P + half * 8);
        acc[nt] = __builtin_amdgcn_wmma_f32_16x16x32_bf16(
            false, a, false, bb[nt % 3], (short)0, acc[nt], false, false);
    }
}

// async-copy one 64B weight row chunk (row n, 32 bf16) global -> LDS (ASYNCcnt path)
__device__ __forceinline__ void async_stage_row(const __bf16* g, __bf16* l) {
    unsigned long long ga = (unsigned long long)(uintptr_t)g;
    unsigned la = (unsigned)(uintptr_t)l;   // low 32 bits of flat LDS address = LDS offset
    asm volatile("global_load_async_to_lds_b128 %0, %1, off\n\t"
                 "global_load_async_to_lds_b128 %0, %1, off offset:16\n\t"
                 "global_load_async_to_lds_b128 %0, %1, off offset:32\n\t"
                 "global_load_async_to_lds_b128 %0, %1, off offset:48"
                 :: "v"(la), "v"(ga));
}

__device__ __forceinline__ void wait_async_all() {
    asm volatile("s_wait_asynccnt 0x0" ::: "memory");
}

__global__ void __launch_bounds__(256)
mlp_kernel(const float* __restrict__ xc,
           const __bf16* __restrict__ W1b, const float* __restrict__ b1,
           const __bf16* __restrict__ W2b, const float* __restrict__ b2,
           const __bf16* __restrict__ W3b, const float* __restrict__ b3,
           float* __restrict__ out) {
    __shared__ __bf16 wsl[2][256 * SLICE_P];      // double-buffered weight K-slice (block-shared)
    __shared__ __bf16 tiles[4][16 * TILE_P];      // per-wave-pair activation transpose tiles
    const int tid  = threadIdx.x;
    const int wv   = tid >> 5;
    const int g    = wv >> 1;       // row-group (pair) id: 4 groups of 16 rows
    const int nh   = wv & 1;        // which half of the N range this wave owns
    const int lane = tid & 31;
    const int half = lane >> 4;
    const int ln   = lane & 15;
    const int rowBase = blockIdx.x * 64 + g * 16;
    __bf16* tile = &tiles[g][0];                  // shared by the wave pair
    const __bf16* a_ldsrow = tile + ln * TILE_P;

    int rm = rowBase + ln;
    if (rm >= NN) rm = NN - 1;                    // clamp loads, keep EXEC all-1 for WMMA
    const float* arow = xc + (long long)rm * CAT;

    const int nbase = nh * 128;                   // layers 1/2: this wave's column base

    v8f zero = {};
    v8f acc[8];

    // ======== Layer 1: [16 x 1024] x W1^T -> [16 x 256], relu ========
#pragma unroll
    for (int nt = 0; nt < 8; ++nt) acc[nt] = zero;

    async_stage_row(W1b + (long long)tid * CAT, &wsl[0][tid * SLICE_P]);
    wait_async_all();
    __syncthreads();

    v16bf a = load_a_f32(arow, half * 8);
    for (int ks = 0; ks < 32; ++ks) {
        const __bf16* bs = &wsl[ks & 1][0];
        if (ks + 1 < 32)   // prefetch next weight slice into the other buffer (async DMA)
            async_stage_row(W1b + (long long)tid * CAT + (ks + 1) * 32,
                            &wsl[(ks + 1) & 1][tid * SLICE_P]);
        const int kn = (ks + 1 < 32) ? (ks + 1) : 31;
        v16bf a_next = load_a_f32(arow, kn * 32 + half * 8);   // pipeline A
        wmma_kstep<8>(bs + nbase * SLICE_P, ln, half, a, acc);
        a = a_next;
        wait_async_all();
        __syncthreads();
    }
#pragma unroll
    for (int nt = 0; nt < 8; ++nt) {
        const int nn = nbase + nt * 16 + ln;
        const float bias = b1[nn];
#pragma unroll
        for (int i = 0; i < 8; ++i) {
            float v = acc[nt][i] + bias;
            tile[(i + 8 * half) * TILE_P + nn] = (__bf16)fmaxf(v, 0.0f);
        }
    }

    // ======== Layer 2: [16 x 256] x W2^T -> [16 x 256], relu ========
#pragma unroll
    for (int nt = 0; nt < 8; ++nt) acc[nt] = zero;

    async_stage_row(W2b + (long long)tid * HIDC, &wsl[0][tid * SLICE_P]);
    wait_async_all();
    __syncthreads();   // also makes partner's layer-1 tile columns visible

    for (int ks = 0; ks < 8; ++ks) {
        const __bf16* bs = &wsl[ks & 1][0];
        if (ks + 1 < 8)
            async_stage_row(W2b + (long long)tid * HIDC + (ks + 1) * 32,
                            &wsl[(ks + 1) & 1][tid * SLICE_P]);
        v16bf a2 = load_frag_bf16(a_ldsrow + ks * 32 + half * 8);
        wmma_kstep<8>(bs + nbase * SLICE_P, ln, half, a2, acc);
        wait_async_all();
        __syncthreads();
    }
#pragma unroll
    for (int nt = 0; nt < 8; ++nt) {
        const int nn = nbase + nt * 16 + ln;
        const float bias = b2[nn];
#pragma unroll
        for (int i = 0; i < 8; ++i) {
            float v = acc[nt][i] + bias;
            tile[(i + 8 * half) * TILE_P + nn] = (__bf16)fmaxf(v, 0.0f);
        }
    }

    // ======== Layer 3: [16 x 256] x W3^T -> [16 x 128] (64 cols per wave) ========
#pragma unroll
    for (int nt = 0; nt < 4; ++nt) acc[nt] = zero;

    if (tid < OUTC)
        async_stage_row(W3b + (long long)tid * HIDC, &wsl[0][tid * SLICE_P]);
    wait_async_all();
    __syncthreads();   // also makes partner's layer-2 tile columns visible

    const int nbase3 = nh * 64;
    for (int ks = 0; ks < 8; ++ks) {
        const __bf16* bs = &wsl[ks & 1][0];
        if (ks + 1 < 8 && tid < OUTC)
            async_stage_row(W3b + (long long)tid * HIDC + (ks + 1) * 32,
                            &wsl[(ks + 1) & 1][tid * SLICE_P]);
        v16bf a3 = load_frag_bf16(a_ldsrow + ks * 32 + half * 8);
        wmma_kstep<4>(bs + nbase3 * SLICE_P, ln, half, a3, acc);
        wait_async_all();
        __syncthreads();
    }
#pragma unroll
    for (int nt = 0; nt < 4; ++nt) {
        const int nn = nbase3 + nt * 16 + ln;
        const float bias = b3[nn];
#pragma unroll
        for (int i = 0; i < 8; ++i) {
            int r = rowBase + i + 8 * half;
            if (r < NN) out[(long long)r * OUTC + nn] = acc[nt][i] + bias;
        }
    }
}

// ---------------- launcher ----------------

extern "C" void kernel_launch(void* const* d_in, const int* in_sizes, int n_in,
                              void* d_out, int out_size, void* d_ws, size_t ws_size,
                              hipStream_t stream) {
    const float*     x   = (const float*)d_in[0];
    const long long* ei  = (const long long*)d_in[1];
    const float*     W1  = (const float*)d_in[2];
    const float*     b1  = (const float*)d_in[3];
    const float*     W2  = (const float*)d_in[4];
    const float*     b2  = (const float*)d_in[5];
    const float*     W3  = (const float*)d_in[6];
    const float*     b3  = (const float*)d_in[7];
    float* out = (float*)d_out;

    const long long* rowI = ei;
    const long long* colI = ei + NE;

    char* p = (char*)d_ws;
    float* xc   = (float*)p;  p += sizeof(float) * (size_t)NN * CAT;      // 409.6 MB
    float* val  = (float*)p;  p += sizeof(float) * (size_t)NE;            // 12.8 MB
    float* deg  = (float*)p;  p += sizeof(float) * (size_t)NN;            // 0.4 MB
    __bf16* W1bf = (__bf16*)p; p += sizeof(__bf16) * (size_t)HIDC * CAT;
    __bf16* W2bf = (__bf16*)p; p += sizeof(__bf16) * (size_t)HIDC * HIDC;
    __bf16* W3bf = (__bf16*)p; p += sizeof(__bf16) * (size_t)OUTC * HIDC;

    const int TB = 256;
    const long long totalXC = (long long)NN * CAT;

    init_xc_kernel<<<(unsigned)((totalXC + TB - 1) / TB), TB, 0, stream>>>(x, xc, totalXC);
    zero_f32_kernel<<<(NN + TB - 1) / TB, TB, 0, stream>>>(deg, NN);
    cvt_bf16_kernel<<<(HIDC * CAT + TB - 1) / TB, TB, 0, stream>>>(W1, W1bf, HIDC * CAT);
    cvt_bf16_kernel<<<(HIDC * HIDC + TB - 1) / TB, TB, 0, stream>>>(W2, W2bf, HIDC * HIDC);
    cvt_bf16_kernel<<<(OUTC * HIDC + TB - 1) / TB, TB, 0, stream>>>(W3, W3bf, OUTC * HIDC);
    degree_kernel<<<(NE + TB - 1) / TB, TB, 0, stream>>>(rowI, deg, NE);
    rsqrt_kernel<<<(NN + TB - 1) / TB, TB, 0, stream>>>(deg, NN);
    edge_value_kernel<<<(NE + TB - 1) / TB, TB, 0, stream>>>(rowI, colI, deg, val, NE);

    const long long spmmThreads = (long long)NE * 32;
    const unsigned  spmmBlocks  = (unsigned)((spmmThreads + TB - 1) / TB);
    for (int hop = 0; hop < 3; ++hop) {
        spmm_kernel<<<spmmBlocks, TB, 0, stream>>>(rowI, colI, val, xc,
                                                   hop * INC, (hop + 1) * INC, NE);
    }

    mlp_kernel<<<(NN + 63) / 64, TB, 0, stream>>>(xc, W1bf, b1, W2bf, b2, W3bf, b3, out);
}
